// mambaModel_49452253446253
// MI455X (gfx1250) — compile-verified, matrix operand order checked by reading, not run on hardware
//
#include <hip/hip_runtime.h>
#include <hip/hip_bf16.h>
#include <math.h>

// ---------------- model dims ----------------
#define BATCH     64
#define INPUT_DIM 32768
#define D_MODEL   1024
#define D_INNER   2048
#define D_STATE   16
#define DT_RANK   64
#define SEQ_L     32              // mamba sequence length (= 32 conv channels)
#define MROWS     (BATCH * SEQ_L) // 2048 GEMM rows
#define XD_COLS   96              // dt_rank + 2*d_state
#define XD_PAD    128             // padded to a multiple of BN for unguarded GEMM

typedef __attribute__((ext_vector_type(16))) __bf16 v16bf;
typedef __attribute__((ext_vector_type(8)))  __bf16 v8bf;
typedef __attribute__((ext_vector_type(4)))  __bf16 v4bf;
typedef __attribute__((ext_vector_type(8)))  float  v8f;

union Frag { v16bf v; v8bf h[2]; };

// =====================================================================
// fp32-in / bf16-WMMA / fp32-out GEMM:  C = A(MxK) * B(KxN) [+ R]
//   TBM x 128 block tile, BK=32, 256 threads (8 waves).
//   TBM=128: each wave owns 2x4 sub-tiles -> 8 WMMA per K-step.
//   TBM=64 : each wave owns 1x4 sub-tiles (used for the small FC).
//   Register double-buffering: next K-tile global loads issued before the
//   WMMA phase so latency hides behind matrix math.
//   REQUIRES: M%TBM==0, N%128==0, K%32==0, strides multiple of 4 floats.
//   LDS rows padded to 80B (20 banks): conflict-free 16-lane b128 reads.
// =====================================================================
#define BN 128
#define BK 32
#define LDP (BK + 8)   // padded row length in bf16 elements (80 bytes)

template <int TBM>
__global__ __launch_bounds__(256)
void gemm_wmma_bf16(const float* __restrict__ A, int lda,
                    const float* __restrict__ Bw, int ldb,
                    float* __restrict__ C, int ldc,
                    const float* __restrict__ Rez,   // optional residual (nullptr = none)
                    int K)
{
    constexpr int RT  = TBM / 64;                 // row sub-tiles per wave (1 or 2)
    constexpr int AIT = (TBM * BK / 4) / 256;     // float4 A loads per thread
    constexpr int BIT = (BK * BN / 4) / 256;      // float4 B loads per thread (4)

    __shared__ __bf16 As[TBM][LDP];   // A tile, row-major (m, k)
    __shared__ __bf16 Bs[BN][LDP];    // B tile, TRANSPOSED (n, k)

    const int tile_m = blockIdx.x * TBM;
    const int tile_n = blockIdx.y * BN;
    const int tid  = threadIdx.x;
    const int wave = tid >> 5;
    const int lane = tid & 31;
    const int kh   = lane >> 4;        // lane half (0/1)
    const int l16  = lane & 15;

    const int sm = (wave & 3) * (16 * RT);  // wave's row sub-tiles: sm + rt*16
    const int sn = (wave >> 2) * 64;        // wave's col sub-tiles: sn + {0,16,32,48}

    v8f acc[RT][4] = {};

    // ---- register-staged K tile (double buffer vs. LDS) ----
    float4 ra[AIT], rb[BIT];
#pragma unroll
    for (int it = 0; it < AIT; ++it) {
        int idx = tid + it * 256;
        int r = idx >> 3, c4 = idx & 7;
        ra[it] = *(const float4*)&A[(size_t)(tile_m + r) * lda + c4 * 4];
    }
#pragma unroll
    for (int it = 0; it < BIT; ++it) {
        int idx = tid + it * 256;
        int r = idx >> 5, c4 = idx & 31;
        rb[it] = *(const float4*)&Bw[(size_t)r * ldb + tile_n + c4 * 4];
    }

    for (int k0 = 0; k0 < K; k0 += BK) {
        // ---- convert staged registers -> LDS ----
#pragma unroll
        for (int it = 0; it < AIT; ++it) {
            int idx = tid + it * 256;
            int r = idx >> 3, c4 = idx & 7;
            v4bf hv;
            hv[0] = (__bf16)ra[it].x; hv[1] = (__bf16)ra[it].y;
            hv[2] = (__bf16)ra[it].z; hv[3] = (__bf16)ra[it].w;
            *(v4bf*)&As[r][c4 * 4] = hv;
        }
#pragma unroll
        for (int it = 0; it < BIT; ++it) {
            int idx = tid + it * 256;
            int r = idx >> 5, c4 = idx & 31;    // r = k, cols c4*4..+3 (transpose scatter)
            Bs[c4 * 4 + 0][r] = (__bf16)rb[it].x;
            Bs[c4 * 4 + 1][r] = (__bf16)rb[it].y;
            Bs[c4 * 4 + 2][r] = (__bf16)rb[it].z;
            Bs[c4 * 4 + 3][r] = (__bf16)rb[it].w;
        }
        __syncthreads();

        // ---- issue next K-tile global loads (latency hidden by WMMAs) ----
        if (k0 + BK < K) {
            const int kn = k0 + BK;
#pragma unroll
            for (int it = 0; it < AIT; ++it) {
                int idx = tid + it * 256;
                int r = idx >> 3, c4 = idx & 7;
                ra[it] = *(const float4*)&A[(size_t)(tile_m + r) * lda + kn + c4 * 4];
            }
#pragma unroll
            for (int it = 0; it < BIT; ++it) {
                int idx = tid + it * 256;
                int r = idx >> 5, c4 = idx & 31;
                rb[it] = *(const float4*)&Bw[(size_t)(kn + r) * ldb + tile_n + c4 * 4];
            }
        }

        // ---- fragments: two contiguous 16B chunks per lane (2x ds_load_b128) ----
        Frag fa[RT];
#pragma unroll
        for (int rt = 0; rt < RT; ++rt) {
            const __bf16* ar = &As[sm + rt * 16 + l16][0];
            fa[rt].h[0] = *(const v8bf*)(ar + kh * 8);        // K kh*8 .. +7
            fa[rt].h[1] = *(const v8bf*)(ar + 16 + kh * 8);   // K 16+kh*8 .. +7
        }
#pragma unroll
        for (int t = 0; t < 4; ++t) {
            Frag fb;
            const __bf16* br = &Bs[sn + t * 16 + l16][0];
            fb.h[0] = *(const v8bf*)(br + kh * 8);
            fb.h[1] = *(const v8bf*)(br + 16 + kh * 8);
#pragma unroll
            for (int rt = 0; rt < RT; ++rt)
                acc[rt][t] = __builtin_amdgcn_wmma_f32_16x16x32_bf16(
                    false, fa[rt].v, false, fb.v, (short)0, acc[rt][t], false, false);
        }
        __syncthreads();
    }

    // ---- store C (f32 16x16 layout: VGPR i -> M = i + 8*kh, N = lane&15) ----
#pragma unroll
    for (int rt = 0; rt < RT; ++rt) {
#pragma unroll
        for (int t = 0; t < 4; ++t) {
#pragma unroll
            for (int i = 0; i < 8; ++i) {
                int gm = tile_m + sm + rt * 16 + kh * 8 + i;
                int gn = tile_n + sn + t * 16 + l16;
                float r = acc[rt][t][i];
                if (Rez) r += Rez[(size_t)gm * ldc + gn];
                C[(size_t)gm * ldc + gn] = r;
            }
        }
    }
}

// =====================================================================
// zero-pad x_proj_w (2048x96) -> (2048x128)
// =====================================================================
__global__ __launch_bounds__(256)
void pad_xproj_kernel(const float* __restrict__ w, float* __restrict__ wp)
{
    int idx = blockIdx.x * blockDim.x + threadIdx.x;
    if (idx >= D_INNER * XD_PAD) return;
    int c = idx % XD_PAD, r = idx / XD_PAD;
    wp[idx] = (c < XD_COLS) ? w[r * XD_COLS + c] : 0.f;
}

// =====================================================================
// conv1 (stride 16, K=64, pad 24) + ReLU + avgpool(2) : x(B,32768) -> (B,32,1024)
// =====================================================================
__global__ __launch_bounds__(256)
void conv1_pool_kernel(const float* __restrict__ x, const float* __restrict__ w,
                       const float* __restrict__ b, float* __restrict__ out)
{
    int idx = blockIdx.x * blockDim.x + threadIdx.x;
    if (idx >= BATCH * 32 * 1024) return;
    int p  = idx % 1024;
    int c  = (idx / 1024) % 32;
    int bb = idx / (1024 * 32);
    const float* xb = x + (size_t)bb * INPUT_DIM;
    const float* wc = w + c * 64;
    float bias = b[c];
    float acc = 0.f;
#pragma unroll
    for (int t2 = 0; t2 < 2; ++t2) {
        int t = 2 * p + t2;
        float s = bias;
        int start = t * 16 - 24;
        for (int k = 0; k < 64; ++k) {
            int xi = start + k;
            float xv = (xi >= 0 && xi < INPUT_DIM) ? xb[xi] : 0.f;
            s = fmaf(xv, wc[k], s);
        }
        acc += fmaxf(s, 0.f);
    }
    out[idx] = acc * 0.5f;
}

// =====================================================================
// depthwise causal conv (K=4) over L + SiLU : xz(B*L,4096)[:, :2048] -> xm(B*L,2048)
// =====================================================================
__global__ __launch_bounds__(256)
void dwconv_silu_kernel(const float* __restrict__ xz, const float* __restrict__ w,
                        const float* __restrict__ b, float* __restrict__ xm)
{
    int idx = blockIdx.x * blockDim.x + threadIdx.x;
    if (idx >= MROWS * D_INNER) return;
    int d  = idx % D_INNER;
    int l  = (idx / D_INNER) % SEQ_L;
    int bb = idx / (D_INNER * SEQ_L);
    float s = b[d];
#pragma unroll
    for (int k = 0; k < 4; ++k) {
        int li = l + k - 3;
        if (li >= 0)
            s = fmaf(xz[(size_t)(bb * SEQ_L + li) * (2 * D_INNER) + d], w[d * 4 + k], s);
    }
    float sig = 1.f / (1.f + __expf(-s));
    xm[idx] = s * sig;
}

// =====================================================================
// delta = softplus(delta + dt_bias)
// =====================================================================
__global__ __launch_bounds__(256)
void softplus_bias_kernel(float* __restrict__ dt, const float* __restrict__ bias)
{
    int idx = blockIdx.x * blockDim.x + threadIdx.x;
    if (idx >= MROWS * D_INNER) return;
    float v = dt[idx] + bias[idx % D_INNER];
    dt[idx] = (v > 20.f) ? v : log1pf(__expf(v));
}

// =====================================================================
// selective scan: one thread per (b,d); y written in place of xm, gated by silu(z)
// =====================================================================
__global__ __launch_bounds__(256)
void scan_kernel(const float* __restrict__ delta, float* __restrict__ xm_y,
                 const float* __restrict__ xdbl, const float* __restrict__ xz,
                 const float* __restrict__ A_log, const float* __restrict__ Dp)
{
    int idx = blockIdx.x * blockDim.x + threadIdx.x;
    if (idx >= BATCH * D_INNER) return;
    int d  = idx % D_INNER;
    int bb = idx / D_INNER;

    float A[D_STATE];
#pragma unroll
    for (int s = 0; s < D_STATE; ++s) A[s] = -__expf(A_log[d * D_STATE + s]);
    float h[D_STATE] = {0.f};
    float Dd = Dp[d];

    for (int l = 0; l < SEQ_L; ++l) {
        size_t row = (size_t)bb * SEQ_L + l;
        float dl = delta[row * D_INNER + d];
        float xv = xm_y[row * D_INNER + d];
        const float* bc = xdbl + row * XD_PAD;     // padded x_dbl row
        float y = 0.f;
#pragma unroll
        for (int s = 0; s < D_STATE; ++s) {
            float dA = __expf(dl * A[s]);
            h[s] = fmaf(dA, h[s], dl * bc[DT_RANK + s] * xv);
            y = fmaf(h[s], bc[DT_RANK + D_STATE + s], y);
        }
        y = fmaf(xv, Dd, y);
        float z = xz[row * (2 * D_INNER) + D_INNER + d];
        float sz = z / (1.f + __expf(-z));
        xm_y[row * D_INNER + d] = y * sz;
    }
}

// =====================================================================
// conv2 (32ch->32ch, K=3, pad 1) + ReLU + residual : h(B,32,1024) -> out(B,32,1024)
// =====================================================================
__global__ __launch_bounds__(256)
void conv2_relu_res_kernel(const float* __restrict__ h, const float* __restrict__ w,
                           const float* __restrict__ b, float* __restrict__ out)
{
    int idx = blockIdx.x * blockDim.x + threadIdx.x;
    if (idx >= BATCH * 32 * D_MODEL) return;
    int m  = idx % D_MODEL;
    int co = (idx / D_MODEL) % 32;
    int bb = idx / (D_MODEL * 32);
    float s = b[co];
    for (int ci = 0; ci < 32; ++ci) {
        const float* hr = h + ((size_t)bb * 32 + ci) * D_MODEL;
        const float* wr = w + (co * 32 + ci) * 3;
        if (m > 0)            s = fmaf(hr[m - 1], wr[0], s);
        s = fmaf(hr[m], wr[1], s);
        if (m < D_MODEL - 1)  s = fmaf(hr[m + 1], wr[2], s);
    }
    out[idx] = fmaxf(s, 0.f) + h[((size_t)bb * 32 + co) * D_MODEL + m];
}

// =====================================================================
// LayerNorm over last dim (1024), one block per (b,ch) row
// =====================================================================
__global__ __launch_bounds__(256)
void layernorm_kernel(const float* __restrict__ x, const float* __restrict__ g,
                      const float* __restrict__ beta, float* __restrict__ out)
{
    __shared__ float red[256];
    int row = blockIdx.x;                 // b*32 + ch
    const float* xr = x + (size_t)row * D_MODEL;

    float s = 0.f;
    for (int i = threadIdx.x; i < D_MODEL; i += 256) s += xr[i];
    red[threadIdx.x] = s; __syncthreads();
    for (int st = 128; st > 0; st >>= 1) {
        if (threadIdx.x < st) red[threadIdx.x] += red[threadIdx.x + st];
        __syncthreads();
    }
    float mu = red[0] * (1.f / D_MODEL);
    __syncthreads();

    float v = 0.f;
    for (int i = threadIdx.x; i < D_MODEL; i += 256) { float d = xr[i] - mu; v = fmaf(d, d, v); }
    red[threadIdx.x] = v; __syncthreads();
    for (int st = 128; st > 0; st >>= 1) {
        if (threadIdx.x < st) red[threadIdx.x] += red[threadIdx.x + st];
        __syncthreads();
    }
    float rstd = rsqrtf(red[0] * (1.f / D_MODEL) + 1e-5f);

    for (int i = threadIdx.x; i < D_MODEL; i += 256)
        out[(size_t)row * D_MODEL + i] = (xr[i] - mu) * rstd * g[i] + beta[i];
}

// mean over the 32 channel rows -> (B, 1024)
__global__ __launch_bounds__(256)
void chmean_kernel(const float* __restrict__ x, float* __restrict__ out)
{
    int idx = blockIdx.x * blockDim.x + threadIdx.x;
    if (idx >= BATCH * D_MODEL) return;
    int m  = idx % D_MODEL;
    int bb = idx / D_MODEL;
    float s = 0.f;
    for (int c = 0; c < 32; ++c) s += x[((size_t)bb * 32 + c) * D_MODEL + m];
    out[idx] = s * (1.f / 32.f);
}

__global__ __launch_bounds__(256)
void bias_add_kernel(float* __restrict__ out, const float* __restrict__ b, int n, int cols)
{
    int idx = blockIdx.x * blockDim.x + threadIdx.x;
    if (idx < n) out[idx] += b[idx % cols];
}

// =====================================================================
// host-side orchestration
// =====================================================================
extern "C" void kernel_launch(void* const* d_in, const int* in_sizes, int n_in,
                              void* d_out, int out_size, void* d_ws, size_t ws_size,
                              hipStream_t stream)
{
    const float* input_seq = (const float*)d_in[0];
    const float* conv1_w   = (const float*)d_in[1];
    const float* conv1_b   = (const float*)d_in[2];
    const float* conv2_w   = (const float*)d_in[3];
    const float* conv2_b   = (const float*)d_in[4];
    const float* in_proj_w = (const float*)d_in[5];
    const float* convm_w   = (const float*)d_in[6];
    const float* convm_b   = (const float*)d_in[7];
    const float* x_proj_w  = (const float*)d_in[8];
    const float* dt_proj_w = (const float*)d_in[9];
    const float* dt_proj_b = (const float*)d_in[10];
    const float* A_log     = (const float*)d_in[11];
    const float* Dvec      = (const float*)d_in[12];
    const float* out_proj_w= (const float*)d_in[13];
    const float* ln_g      = (const float*)d_in[14];
    const float* ln_b      = (const float*)d_in[15];
    const float* fc_w      = (const float*)d_in[16];
    const float* fc_b      = (const float*)d_in[17];
    float* out = (float*)d_out;

    float* ws     = (float*)d_ws;
    float* bufH   = ws;                                   // 2048*1024 (conv1 out / residual)
    float* bufXZ  = bufH   + (size_t)MROWS * D_MODEL;     // 2048*4096
    float* bufXM  = bufXZ  + (size_t)MROWS * 2 * D_INNER; // 2048*2048 (xm, then y in place)
    float* bufXD  = bufXM  + (size_t)MROWS * D_INNER;     // 2048*128  (padded x_dbl)
    float* bufDT  = bufXD  + (size_t)MROWS * XD_PAD;      // 2048*2048 (delta)
    float* bufG   = bufDT  + (size_t)MROWS * D_INNER;     // 2048*1024 (mamba out + res)
    float* bufOUT = bufG   + (size_t)MROWS * D_MODEL;     // 2048*1024 (layer output)
    float* bufP   = bufOUT + (size_t)MROWS * D_MODEL;     // 64*1024   (pooled)
    float* bufXPW = bufP   + (size_t)BATCH * D_MODEL;     // 2048*128  (padded x_proj_w)

    const int T = 256;
    const int nConv1 = BATCH * 32 * 1024;
    const int nElem  = MROWS * D_INNER;
    const int nScan  = BATCH * D_INNER;

    // zero-pad x_proj_w once per launch
    pad_xproj_kernel<<<(D_INNER * XD_PAD + T - 1) / T, T, 0, stream>>>(x_proj_w, bufXPW);

    for (int layer = 0; layer < 4; ++layer) {
        const float* lin = (layer == 0) ? input_seq : bufOUT;

        conv1_pool_kernel<<<(nConv1 + T - 1) / T, T, 0, stream>>>(lin, conv1_w, conv1_b, bufH);

        // xz = h @ in_proj_w   (2048x1024 * 1024x4096)
        gemm_wmma_bf16<128><<<dim3(MROWS / 128, (2 * D_INNER) / BN), T, 0, stream>>>(
            bufH, D_MODEL, in_proj_w, 2 * D_INNER, bufXZ, 2 * D_INNER, nullptr, D_MODEL);

        dwconv_silu_kernel<<<(nElem + T - 1) / T, T, 0, stream>>>(bufXZ, convm_w, convm_b, bufXM);

        // x_dbl = xm @ x_proj_w_padded  (2048x2048 * 2048x128)
        gemm_wmma_bf16<128><<<dim3(MROWS / 128, XD_PAD / BN), T, 0, stream>>>(
            bufXM, D_INNER, bufXPW, XD_PAD, bufXD, XD_PAD, nullptr, D_INNER);

        // delta = x_dbl[:, :64] @ dt_proj_w  (2048x64 * 64x2048)
        gemm_wmma_bf16<128><<<dim3(MROWS / 128, D_INNER / BN), T, 0, stream>>>(
            bufXD, XD_PAD, dt_proj_w, D_INNER, bufDT, D_INNER, nullptr, DT_RANK);

        softplus_bias_kernel<<<(nElem + T - 1) / T, T, 0, stream>>>(bufDT, dt_proj_b);

        scan_kernel<<<(nScan + T - 1) / T, T, 0, stream>>>(bufDT, bufXM, bufXD, bufXZ, A_log, Dvec);

        // h2 = y @ out_proj_w + h  (2048x2048 * 2048x1024, fused residual)
        gemm_wmma_bf16<128><<<dim3(MROWS / 128, D_MODEL / BN), T, 0, stream>>>(
            bufXM, D_INNER, out_proj_w, D_MODEL, bufG, D_MODEL, bufH, D_INNER);

        conv2_relu_res_kernel<<<(nConv1 + T - 1) / T, T, 0, stream>>>(bufG, conv2_w, conv2_b, bufOUT);
    }

    // LayerNorm over d_model, mean over channels, FC
    layernorm_kernel<<<BATCH * 32, T, 0, stream>>>(bufOUT, ln_g, ln_b, bufG);
    chmean_kernel<<<(BATCH * D_MODEL + T - 1) / T, T, 0, stream>>>(bufG, bufP);

    gemm_wmma_bf16<64><<<dim3(1, 1), T, 0, stream>>>(
        bufP, D_MODEL, fc_w, 128, out, 128, nullptr, D_MODEL);
    bias_add_kernel<<<(BATCH * 128 + T - 1) / T, T, 0, stream>>>(out, fc_b, BATCH * 128, 128);
}
